// GAT_layer_19000935317636
// MI455X (gfx1250) — compile-verified, matrix operand order checked by reading, not run on hardware
//
#include <hip/hip_runtime.h>
#include <hip/hip_bf16.h>

typedef __attribute__((ext_vector_type(16))) _Float16 v16h;
typedef __attribute__((ext_vector_type(8)))  _Float16 v8h;
typedef __attribute__((ext_vector_type(8)))  float    v8f;

#define N_NODES 50000
#define N_EDGES 250000
#define D 128

// ---------------------------------------------------------------------------
// Fragment loaders matching CDNA5 16-bit A/B WMMA VGPR layout (wave32):
// lane l: elems 0..7 = K khalf+0..7, elems 8..15 = K khalf+16..23,
// khalf = (l>>4)*8; A row / B col = l&15.
// ---------------------------------------------------------------------------
__device__ __forceinline__ v16h load_a_frag_f32(const float* __restrict__ p) {
  float4 x0 = *(const float4*)(p + 0);
  float4 x1 = *(const float4*)(p + 4);
  float4 x2 = *(const float4*)(p + 16);
  float4 x3 = *(const float4*)(p + 20);
  v16h f;
  f[0]=(_Float16)x0.x; f[1]=(_Float16)x0.y; f[2]=(_Float16)x0.z; f[3]=(_Float16)x0.w;
  f[4]=(_Float16)x1.x; f[5]=(_Float16)x1.y; f[6]=(_Float16)x1.z; f[7]=(_Float16)x1.w;
  f[8]=(_Float16)x2.x; f[9]=(_Float16)x2.y; f[10]=(_Float16)x2.z; f[11]=(_Float16)x2.w;
  f[12]=(_Float16)x3.x; f[13]=(_Float16)x3.y; f[14]=(_Float16)x3.z; f[15]=(_Float16)x3.w;
  return f;
}

__device__ __forceinline__ v16h load_b_frag_lds(const _Float16* __restrict__ p) {
  v8h lo = *(const v8h*)(p);
  v8h hi = *(const v8h*)(p + 16);
  v16h f;
#pragma unroll
  for (int j = 0; j < 8; ++j) { f[j] = lo[j]; f[8 + j] = hi[j]; }
  return f;
}

// ---------------------------------------------------------------------------
// GEMM: Out[M x 128] = relu?( concat(A0[,A1])[M x K] @ W[K x 128] )
// optionally s_out[M] = Out @ w_score   (K = NUM_A*128)
// 256 threads = 8 waves; each wave: 16-row strip x 128 cols via 16x16x32 WMMA.
// W staged transposed as f16 in LDS: Wt[n*(K+PAD) + k], PAD=8 f16 (+4 banks
// per column -> conflict-free ds_load_b128 fragment reads).
// ---------------------------------------------------------------------------
template <int NUM_A, bool RELU, bool SCORE>
__global__ void __launch_bounds__(256)
gemm_wmma_kernel(const float* __restrict__ A0, const float* __restrict__ A1,
                 const float* __restrict__ W, const float* __restrict__ w_score,
                 float* __restrict__ Out, float* __restrict__ s_out, int M) {
  constexpr int K   = NUM_A * 128;
  constexpr int KT  = K / 32;
  constexpr int PAD = 8;                // f16 elements
  constexpr int LDW = K + PAD;          // LDS row stride (f16)

  extern __shared__ char smem_raw[];
  _Float16* Wt = (_Float16*)smem_raw;

  const int tid = threadIdx.x;
  // Stage W (K x 128 row-major f32) transposed into LDS as f16.
  // One thread per (k8, n) chunk: 8 global b32 loads (coalesced across n),
  // 4 cvt_pk, one ds_store_b128.
  {
    constexpr int NCHUNK = 128 * K / 8;
#pragma unroll
    for (int idx = tid; idx < NCHUNK; idx += 256) {
      const int n  = idx & 127;
      const int k8 = (idx >> 7) << 3;
      v8h h;
#pragma unroll
      for (int j = 0; j < 8; ++j) h[j] = (_Float16)W[(size_t)(k8 + j) * 128 + n];
      *(v8h*)(Wt + (size_t)n * LDW + k8) = h;
    }
  }
  __syncthreads();

  const int wave  = tid >> 5;
  const int lane  = tid & 31;
  const int m     = lane & 15;
  const int khalf = (lane >> 4) << 3;
  const int r0    = blockIdx.x * 128 + wave * 16;

  int rowA = r0 + m;
  if (rowA >= M) rowA = M - 1;  // clamp (stores are guarded / fast-pathed)

  v16h afrag[KT];
#pragma unroll
  for (int kt = 0; kt < KT; ++kt) {
    const float* Asrc = (NUM_A == 2 && kt >= 4) ? A1 : A0;
    const int kb = (NUM_A == 2 && kt >= 4) ? (kt - 4) * 32 : kt * 32;
    afrag[kt] = load_a_frag_f32(Asrc + (size_t)rowA * 128 + kb + khalf);
  }

  float spart[8];
#pragma unroll
  for (int v = 0; v < 8; ++v) spart[v] = 0.0f;

  const int  rbase = r0 + ((lane >> 4) << 3);
  const bool full  = (r0 + 16 <= M);    // wave-uniform: whole strip in range

#pragma unroll
  for (int nt = 0; nt < 8; ++nt) {
    v8f acc = {};
    const int col = nt * 16 + m;
    const _Float16* bp = Wt + (size_t)col * LDW + khalf;
#pragma unroll
    for (int kt = 0; kt < KT; ++kt) {
      v16h bfrag = load_b_frag_lds(bp + kt * 32);
      acc = __builtin_amdgcn_wmma_f32_16x16x32_f16(
          false, afrag[kt], false, bfrag, (short)0, acc, false, false);
    }
    const float wcol = SCORE ? w_score[col] : 0.0f;
    if (full) {
#pragma unroll
      for (int v = 0; v < 8; ++v) {
        float val = acc[v];
        if (RELU) val = fmaxf(val, 0.0f);
        Out[(size_t)(rbase + v) * 128 + col] = val;
        if (SCORE) spart[v] += val * wcol;
      }
    } else {
#pragma unroll
      for (int v = 0; v < 8; ++v) {
        float val = acc[v];
        if (RELU) val = fmaxf(val, 0.0f);
        const int ro = rbase + v;
        if (ro < M) Out[(size_t)ro * 128 + col] = val;
        if (SCORE) spart[v] += val * wcol;
      }
    }
  }

  if (SCORE) {
#pragma unroll
    for (int v = 0; v < 8; ++v) {
      float s = spart[v];
      s += __shfl_xor(s, 1, 32);
      s += __shfl_xor(s, 2, 32);
      s += __shfl_xor(s, 4, 32);
      s += __shfl_xor(s, 8, 32);
      if (m == 0) {
        const int ro = rbase + v;
        if (ro < M) s_out[ro] = s;
      }
    }
  }
}

// ---------------------------------------------------------------------------
// Elementwise / scatter kernels
// ---------------------------------------------------------------------------
__device__ __forceinline__ void atomicMaxFloat(float* addr, float val) {
  if (val >= 0.0f) atomicMax((int*)addr, __float_as_int(val));
  else             atomicMin((unsigned int*)addr, (unsigned int)__float_as_int(val));
}

__global__ void init_seg_kernel(float* __restrict__ mbuf, float* __restrict__ num,
                                float* __restrict__ den, int nNodes) {
  const int i = blockIdx.x * 256 + threadIdx.x;
  if (i < nNodes * D) num[i] = 0.0f;
  if (i < nNodes) { mbuf[i] = -1e30f; den[i] = 0.0f; }
}

__global__ void edge_max_kernel(const int* __restrict__ edges,
                                const float* __restrict__ s_n,
                                const float* __restrict__ s_e,
                                float* __restrict__ mbuf, int nEdges) {
  const int e = blockIdx.x * 256 + threadIdx.x;
  if (e >= nEdges) return;
  const int src = edges[2 * e], dst = edges[2 * e + 1];
  atomicMaxFloat(mbuf + dst, fmaxf(s_n[src], s_e[e]));
}

// one wave per edge: gather F_src/F_dst/F_e, write applied_e, scatter num/den
__global__ void __launch_bounds__(256)
edge_scatter_kernel(const int* __restrict__ edges,
                    const float* __restrict__ Fn, const float* __restrict__ Fe,
                    const float* __restrict__ s_n, const float* __restrict__ s_e,
                    const float* __restrict__ mbuf,
                    float* __restrict__ applied_e,
                    float* __restrict__ num, float* __restrict__ den, int nEdges) {
  const int lane = threadIdx.x & 31;
  const int e = blockIdx.x * 8 + (threadIdx.x >> 5);
  if (e >= nEdges) return;
  const int src = edges[2 * e], dst = edges[2 * e + 1];
  const float ss = s_n[src], sd = s_n[dst], se = s_e[e], mm = mbuf[dst];
  // pairwise softmax over (s_src, s_dst)
  const float mx = fmaxf(ss, sd);
  const float p0 = __expf(ss - mx), p1 = __expf(sd - mx);
  const float inv = 1.0f / (p0 + p1);
  const float a0 = p0 * inv, a1 = p1 * inv;
  // segment-softmax weights
  const float en = __expf(ss - mm), ee = __expf(se - mm);

  const float4* Fs4 = (const float4*)(Fn + (size_t)src * D);
  const float4* Fd4 = (const float4*)(Fn + (size_t)dst * D);
  const float4* Fe4 = (const float4*)(Fe + (size_t)e * D);
  float4* Ap4 = (float4*)(applied_e + (size_t)e * D);
  float* nrow = num + (size_t)dst * D;

  const float4 fs = Fs4[lane];
  const float4 fd = Fd4[lane];
  const float4 fe = Fe4[lane];
  float4 ap;
  ap.x = a0 * fs.x + a1 * fd.x;  ap.y = a0 * fs.y + a1 * fd.y;
  ap.z = a0 * fs.z + a1 * fd.z;  ap.w = a0 * fs.w + a1 * fd.w;
  Ap4[lane] = ap;

  const int c = lane * 4;
  atomicAdd(nrow + c + 0, en * fs.x + ee * fe.x);
  atomicAdd(nrow + c + 1, en * fs.y + ee * fe.y);
  atomicAdd(nrow + c + 2, en * fs.z + ee * fe.z);
  atomicAdd(nrow + c + 3, en * fs.w + ee * fe.w);
  if (lane == 0) atomicAdd(den + dst, en + ee);
}

__global__ void nodes_div_kernel(const float* __restrict__ num,
                                 const float* __restrict__ den,
                                 float* __restrict__ applied_n, int nNodes) {
  const int i = blockIdx.x * 256 + threadIdx.x;
  if (i >= nNodes * D) return;
  applied_n[i] = num[i] / den[i >> 7];
}

// ---------------------------------------------------------------------------
extern "C" void kernel_launch(void* const* d_in, const int* in_sizes, int n_in,
                              void* d_out, int out_size, void* d_ws, size_t ws_size,
                              hipStream_t stream) {
  const float* obj         = (const float*)d_in[0];
  const float* pred        = (const float*)d_in[1];
  const int*   edges       = (const int*)  d_in[2];
  const float* W_node      = (const float*)d_in[3];
  const float* W_obj_score = (const float*)d_in[4];
  const float* W_edge      = (const float*)d_in[5];
  const float* W_rel_score = (const float*)d_in[6];
  const float* W_phi_node  = (const float*)d_in[7];
  const float* W_phi_edge  = (const float*)d_in[8];
  const float* W_node_out  = (const float*)d_in[9];
  const float* W_edge_out  = (const float*)d_in[10];
  float* out = (float*)d_out;

  char* wsp = (char*)d_ws;
  auto alloc = [&](size_t nfloats) {
    float* p = (float*)wsp;
    wsp += ((nfloats * sizeof(float) + 255) / 256) * 256;
    return p;
  };
  float* Fn_a      = alloc((size_t)N_NODES * D);
  float* Fn_b      = alloc((size_t)N_NODES * D);
  float* Fn_lin    = alloc((size_t)N_NODES * D);
  float* applied_n = alloc((size_t)N_NODES * D);
  float* numbuf    = alloc((size_t)N_NODES * D);
  float* Fe_a      = alloc((size_t)N_EDGES * D);
  float* Fe_b      = alloc((size_t)N_EDGES * D);
  float* Fe_new    = alloc((size_t)N_EDGES * D);
  float* applied_e = alloc((size_t)N_EDGES * D);
  float* s_n       = alloc(N_NODES);
  float* s_e       = alloc(N_EDGES);
  float* mbuf      = alloc(N_NODES);
  float* denbuf    = alloc(N_NODES);

  const dim3 blk(256);
  const int gN = (N_NODES + 127) / 128;
  const int gE = (N_EDGES + 127) / 128;
  const size_t smK128 = (size_t)128 * (128 + 8) * sizeof(_Float16);  // ~34 KB
  const size_t smK256 = (size_t)128 * (256 + 8) * sizeof(_Float16);  // ~66 KB
  const int gInit = (N_NODES * D + 255) / 256;

  const float* Fn_cur = obj;
  const float* Fe_cur = pred;
  float* Fn_next[2] = {Fn_a, Fn_b};
  float* Fe_next[2] = {Fe_a, Fe_b};

  for (int step = 0; step < 2; ++step) {
    // F_n_lin = F_n @ W_node ; s_n = F_n_lin @ W_obj_score
    gemm_wmma_kernel<1, false, true><<<gN, blk, smK128, stream>>>(
        Fn_cur, nullptr, W_node, W_obj_score, Fn_lin, s_n, N_NODES);
    // F_e_new = F_e @ W_edge ; s_e = F_e_new @ W_rel_score
    gemm_wmma_kernel<1, false, true><<<gE, blk, smK128, stream>>>(
        Fe_cur, nullptr, W_edge, W_rel_score, Fe_new, s_e, N_EDGES);

    init_seg_kernel<<<gInit, blk, 0, stream>>>(mbuf, numbuf, denbuf, N_NODES);
    edge_max_kernel<<<(N_EDGES + 255) / 256, blk, 0, stream>>>(
        edges, s_n, s_e, mbuf, N_EDGES);
    edge_scatter_kernel<<<(N_EDGES + 7) / 8, blk, 0, stream>>>(
        edges, Fn_lin, Fe_new, s_n, s_e, mbuf, applied_e, numbuf, denbuf, N_EDGES);

    // F_e_next = relu([applied_e | F_e_new] @ W_phi_edge)
    gemm_wmma_kernel<2, true, false><<<gE, blk, smK256, stream>>>(
        applied_e, Fe_new, W_phi_edge, nullptr, Fe_next[step], nullptr, N_EDGES);

    nodes_div_kernel<<<gInit, blk, 0, stream>>>(numbuf, denbuf, applied_n, N_NODES);

    // F_n_next = relu([applied_n | F_n_lin] @ W_phi_node)
    gemm_wmma_kernel<2, true, false><<<gN, blk, smK256, stream>>>(
        applied_n, Fn_lin, W_phi_node, nullptr, Fn_next[step], nullptr, N_NODES);

    Fn_cur = Fn_next[step];
    Fe_cur = Fe_next[step];
  }

  // Output projections: nodes first, then edges (tuple return order, flat).
  gemm_wmma_kernel<1, false, false><<<gN, blk, smK128, stream>>>(
      Fn_cur, nullptr, W_node_out, nullptr, out, nullptr, N_NODES);
  gemm_wmma_kernel<1, false, false><<<gE, blk, smK128, stream>>>(
      Fe_cur, nullptr, W_edge_out, nullptr, out + (size_t)N_NODES * D, nullptr, N_EDGES);
}
